// Block_5557687681664
// MI455X (gfx1250) — matrix operand order repaired and checked by reading, not compiled, not run on hardware
//
#include <hip/hip_runtime.h>
#include <hip/hip_bf16.h>

// ---------- types ----------
typedef __attribute__((ext_vector_type(16))) __bf16       v16bf;
typedef __attribute__((ext_vector_type(8)))  float        v8f;
typedef __attribute__((ext_vector_type(4)))  unsigned int v4u;
typedef __attribute__((ext_vector_type(4)))  float        v4f;

union Frag16 { v4u u[2]; v16bf v; };

__device__ __forceinline__ unsigned short f2b(float f) {
    union { float f; unsigned int u; } c; c.f = f;
    unsigned int u = c.u;
    unsigned int r = u + 0x7FFFu + ((u >> 16) & 1u);   // round-to-nearest-even
    return (unsigned short)(r >> 16);
}
__device__ __forceinline__ unsigned int pkbf(float a, float b) {
    return (unsigned int)f2b(a) | ((unsigned int)f2b(b) << 16);
}
// xor-16 lane exchange within wave32 via ds_swizzle (and=0x1F, or=0, xor=0x10)
#define SWZ_XOR16 0x401F
__device__ __forceinline__ float swzf(float x) {
    return __int_as_float(__builtin_amdgcn_ds_swizzle(__float_as_int(x), SWZ_XOR16));
}
__device__ __forceinline__ unsigned int swzu(unsigned int x) {
    return (unsigned int)__builtin_amdgcn_ds_swizzle((int)x, SWZ_XOR16);
}

#define WMMA_BF16(A_, B_, C_) \
    __builtin_amdgcn_wmma_f32_16x16x32_bf16(false, (A_), false, (B_), (short)0, (C_), false, false)

// problem constants
#define SEQ   2048
#define BATCH 2
#define CDIM  1024
#define HEADS 16
#define HDIM  64
#define MLPD  4096
#define MROWS (BATCH*SEQ)   // 4096 tokens

// ---------- weight transpose + f32->bf16 : Wt[n*K + k] = bf16(W[k*N + n]) ----------
__global__ __launch_bounds__(256) void cvtT_kernel(const float* __restrict__ W,
                                                   unsigned short* __restrict__ Wt,
                                                   int K, int N) {
    size_t idx = (size_t)blockIdx.x * 256 + threadIdx.x;
    if (idx >= (size_t)K * N) return;
    int n = (int)(idx / K);
    int k = (int)(idx % K);
    Wt[idx] = f2b(W[(size_t)k * N + n]);
}

// ---------- layernorm (C=1024) -> bf16 ----------
__global__ __launch_bounds__(256) void ln_kernel(const float* __restrict__ X,
                                                 const float* __restrict__ gam,
                                                 const float* __restrict__ bet,
                                                 unsigned short* __restrict__ Hout) {
    __shared__ float red[256];
    const int r = blockIdx.x, t = threadIdx.x;
    v4f xv = *(const v4f*)(X + (size_t)r * CDIM + t * 4);
    float s  = xv[0] + xv[1] + xv[2] + xv[3];
    float ss = xv[0]*xv[0] + xv[1]*xv[1] + xv[2]*xv[2] + xv[3]*xv[3];
    red[t] = s; __syncthreads();
    for (int o = 128; o > 0; o >>= 1) { if (t < o) red[t] += red[t + o]; __syncthreads(); }
    float mean = red[0] * (1.0f / CDIM); __syncthreads();
    red[t] = ss; __syncthreads();
    for (int o = 128; o > 0; o >>= 1) { if (t < o) red[t] += red[t + o]; __syncthreads(); }
    float var  = red[0] * (1.0f / CDIM) - mean * mean;
    float rstd = rsqrtf(var + 1e-5f);
#pragma unroll
    for (int i = 0; i < 4; ++i) {
        int c = t * 4 + i;
        Hout[(size_t)r * CDIM + c] = f2b((xv[i] - mean) * rstd * gam[c] + bet[c]);
    }
}

// ---------- generic bf16 WMMA GEMM: 128x128 tile, BK=32, 8 waves ----------
// EPI: 0 = QKV scatter (Q scaled by 0.125, V stored transposed)
//      1 = f32 out + bias + residual
//      2 = bf16 out + bias + exact GELU
template<int EPI>
__global__ __launch_bounds__(256) void gemm_bf16_kernel(
    const unsigned short* __restrict__ A,   // [M,K] bf16 row-major
    const unsigned short* __restrict__ Bt,  // [Nn,K] bf16 (B transposed)
    const float* __restrict__ bias,
    const float* __restrict__ resid,
    float* __restrict__ outF,
    unsigned short* __restrict__ outB,
    unsigned short* __restrict__ Qp, unsigned short* __restrict__ Kp,
    unsigned short* __restrict__ Vtp,
    int M, int Nn, int K)
{
    __shared__ alignas(16) unsigned short As[128][32];
    __shared__ alignas(16) unsigned short Bs[128][32];
    const int m0 = blockIdx.y * 128, n0 = blockIdx.x * 128;
    const int tid = threadIdx.x, lane = tid & 31, wid = tid >> 5;
    const int wm = (wid & 3) * 32;          // 4 waves along M (32 rows each)
    const int wn = (wid >> 2) * 64;         // 2 waves along N (64 cols each)
    const int half = lane >> 4, ln = lane & 15;

    v8f acc[2][4] = {};

    for (int k0 = 0; k0 < K; k0 += 32) {
        __syncthreads();
#pragma unroll
        for (int i = 0; i < 2; ++i) {
            int idx = tid * 2 + i;
            int row = idx >> 2, kc = (idx & 3) * 8;
            *(v4u*)&As[row][kc] = *(const v4u*)&A [(size_t)(m0 + row) * K + k0 + kc];
            *(v4u*)&Bs[row][kc] = *(const v4u*)&Bt[(size_t)(n0 + row) * K + k0 + kc];
        }
        if (k0 + 32 < K) {  // prefetch next K tile (global_prefetch_b8)
            __builtin_prefetch(&A [(size_t)(m0 + (tid >> 1)) * K + k0 + 32], 0, 1);
            __builtin_prefetch(&Bt[(size_t)(n0 + (tid >> 1)) * K + k0 + 32], 0, 1);
        }
        __syncthreads();

        Frag16 af[2];
#pragma unroll
        for (int mi = 0; mi < 2; ++mi) {
            const unsigned short* p = &As[wm + mi * 16 + ln][half * 8];
            af[mi].u[0] = *(const v4u*)p;
            af[mi].u[1] = *(const v4u*)(p + 16);
        }
        Frag16 bfr[4];
#pragma unroll
        for (int ni = 0; ni < 4; ++ni) {
            const unsigned short* p = &Bs[wn + ni * 16 + ln][half * 8];
            bfr[ni].u[0] = *(const v4u*)p;
            bfr[ni].u[1] = *(const v4u*)(p + 16);
        }
#pragma unroll
        for (int mi = 0; mi < 2; ++mi)
#pragma unroll
            for (int ni = 0; ni < 4; ++ni)
                acc[mi][ni] = WMMA_BF16(af[mi].v, bfr[ni].v, acc[mi][ni]);
    }

    // epilogue: C layout -> lane holds rows v (lanes 0-15) / v+8 (lanes 16-31), col = lane%16
#pragma unroll
    for (int mi = 0; mi < 2; ++mi)
#pragma unroll
        for (int ni = 0; ni < 4; ++ni)
#pragma unroll
            for (int v = 0; v < 8; ++v) {
                int gr = m0 + wm + mi * 16 + v + half * 8;
                int gc = n0 + wn + ni * 16 + ln;
                float val = acc[mi][ni][v] + bias[gc];
                if (EPI == 0) {
                    int sec = gc >> 10, c = gc & 1023;
                    int hh = c >> 6, dd = c & 63;
                    int bb = gr >> 11, nn = gr & 2047;
                    if (sec == 0)
                        Qp[(((size_t)bb * HEADS + hh) * SEQ + nn) * HDIM + dd] = f2b(val * 0.125f);
                    else if (sec == 1)
                        Kp[(((size_t)bb * HEADS + hh) * SEQ + nn) * HDIM + dd] = f2b(val);
                    else
                        Vtp[(((size_t)bb * HEADS + hh) * HDIM + dd) * SEQ + nn] = f2b(val);
                } else if (EPI == 1) {
                    outF[(size_t)gr * Nn + gc] = val + resid[(size_t)gr * Nn + gc];
                } else {
                    float g = 0.5f * val * (1.0f + erff(val * 0.70710678118f));
                    outB[(size_t)gr * Nn + gc] = f2b(g);
                }
            }
}

// ---------- flash attention, transposed-S design ----------
// block = (b,h, 128 q rows), wave = 16 q rows; no LDS, no barriers.
// S^T = K * Q^T  -> each lane privately holds 16 of 32 KV scores of ONE q row
// O^T = V^T * P^T -> per-lane scalar softmax state; contiguous 16B output stores
__global__ __launch_bounds__(256) void attn_kernel(
    const unsigned short* __restrict__ Q,   // [B,H,N,64] bf16 (pre-scaled by 1/8)
    const unsigned short* __restrict__ Kc,  // [B,H,N,64]
    const unsigned short* __restrict__ Vt,  // [B,H,64,N]
    unsigned short* __restrict__ Y)         // [B*N, C] bf16
{
    const int lane = threadIdx.x & 31, wid = threadIdx.x >> 5;
    const int half = lane >> 4, ln = lane & 15;
    const int qtile = blockIdx.x & 15;
    const int bh = blockIdx.x >> 4;
    const int b = bh >> 4, h = bh & 15;
    const size_t base = (size_t)bh * SEQ * HDIM;
    const unsigned short* Qh = Q  + base;
    const unsigned short* Kh = Kc + base;
    const unsigned short* Vh = Vt + base;
    const int q0 = qtile * 128 + wid * 16;

    // Q as B-fragments (lane ln = q column; K-dim = d, contiguous)
    Frag16 qb[2];
#pragma unroll
    for (int c = 0; c < 2; ++c) {
        const unsigned short* p = Qh + (size_t)(q0 + ln) * HDIM + c * 32 + half * 8;
        qb[c].u[0] = *(const v4u*)p;
        qb[c].u[1] = *(const v4u*)(p + 16);
    }

    v8f O[4] = {};            // O^T tiles: d in [t*16, t*16+16)
    float m = -3.0e38f, l = 0.0f;   // per-lane (per q row) softmax state

    for (int kv0 = 0; kv0 < SEQ; kv0 += 32) {
        // K as A-fragments (row = kv, K-dim = d)
        Frag16 ka[2][2];
#pragma unroll
        for (int j = 0; j < 2; ++j)
#pragma unroll
            for (int c = 0; c < 2; ++c) {
                const unsigned short* p =
                    Kh + (size_t)(kv0 + j * 16 + ln) * HDIM + c * 32 + half * 8;
                ka[j][c].u[0] = *(const v4u*)p;
                ka[j][c].u[1] = *(const v4u*)(p + 16);
            }
        v8f T0 = {}, T1 = {};   // S^T tiles; lane holds kv {kv0+8h+v} / {kv0+16+8h+v}
        T0 = WMMA_BF16(ka[0][0].v, qb[0].v, T0);
        T0 = WMMA_BF16(ka[0][1].v, qb[1].v, T0);
        T1 = WMMA_BF16(ka[1][0].v, qb[0].v, T1);
        T1 = WMMA_BF16(ka[1][1].v, qb[1].v, T1);

        // online softmax: 15 in-lane max + ONE xor-16 exchange
        float mx = fmaxf(T0[0], T1[0]);
#pragma unroll
        for (int v = 1; v < 8; ++v) mx = fmaxf(mx, fmaxf(T0[v], T1[v]));
        mx = fmaxf(mx, swzf(mx));
        float mnew = fmaxf(m, mx);
        float al = __expf(m - mnew);
        float rs = 0.0f;
#pragma unroll
        for (int v = 0; v < 8; ++v) { T0[v] = __expf(T0[v] - mnew); rs += T0[v]; }
#pragma unroll
        for (int v = 0; v < 8; ++v) { T1[v] = __expf(T1[v] - mnew); rs += T1[v]; }
        rs += swzf(rs);
        l = l * al + rs;
        m = mnew;
#pragma unroll
        for (int t = 0; t < 4; ++t)
#pragma unroll
            for (int v = 0; v < 8; ++v) O[t][v] *= al;

        // build P^T B-fragment: pack own 16 P values to bf16, exchange 8 dwords xor-16
        unsigned int pk0[4], pk1[4], r0[4], r1[4];
#pragma unroll
        for (int d = 0; d < 4; ++d) {
            pk0[d] = pkbf(T0[2 * d], T0[2 * d + 1]);
            pk1[d] = pkbf(T1[2 * d], T1[2 * d + 1]);
        }
#pragma unroll
        for (int d = 0; d < 4; ++d) { r0[d] = swzu(pk0[d]); r1[d] = swzu(pk1[d]); }
        // lanes<16 need K=kv0..kv0+15  = {own T0 (kv 0-7),   partner T0 (kv 8-15)}
        // lanes>=16 need K=kv0+16..+31 = {partner T1 (16-23), own T1 (kv 24-31)}
        Frag16 pb;
        {
            unsigned int w0 = half ? r1[0] : pk0[0], w1 = half ? r1[1] : pk0[1];
            unsigned int w2 = half ? r1[2] : pk0[2], w3 = half ? r1[3] : pk0[3];
            unsigned int w4 = half ? pk1[0] : r0[0], w5 = half ? pk1[1] : r0[1];
            unsigned int w6 = half ? pk1[2] : r0[2], w7 = half ? pk1[3] : r0[3];
            v4u lo = {w0, w1, w2, w3}, hi = {w4, w5, w6, w7};
            pb.u[0] = lo; pb.u[1] = hi;
        }

        // O^T += V^T * P^T  (V^T rows are contiguous in Vt[d][n])
#pragma unroll
        for (int t = 0; t < 4; ++t) {
            Frag16 va;
            const unsigned short* p = Vh + (size_t)(t * 16 + ln) * SEQ + kv0 + half * 16;
            va.u[0] = *(const v4u*)p;
            va.u[1] = *(const v4u*)(p + 8);
            O[t] = WMMA_BF16(va.v, pb.v, O[t]);
        }
    }

    // finalize: per-lane scale, pack 8 consecutive bf16, one 16B store per tile
    float inv = 1.0f / l;
    const size_t yb = ((size_t)b * SEQ + q0 + ln) * CDIM + h * HDIM + half * 8;
#pragma unroll
    for (int t = 0; t < 4; ++t) {
        unsigned int o0 = pkbf(O[t][0] * inv, O[t][1] * inv);
        unsigned int o1 = pkbf(O[t][2] * inv, O[t][3] * inv);
        unsigned int o2 = pkbf(O[t][4] * inv, O[t][5] * inv);
        unsigned int o3 = pkbf(O[t][6] * inv, O[t][7] * inv);
        v4u st = {o0, o1, o2, o3};
        *(v4u*)&Y[yb + t * 16] = st;
    }
}

// ---------- host side ----------
extern "C" void kernel_launch(void* const* d_in, const int* in_sizes, int n_in,
                              void* d_out, int out_size, void* d_ws, size_t ws_size,
                              hipStream_t stream) {
    const float* x      = (const float*)d_in[0];
    const float* ln1_g  = (const float*)d_in[1];
    const float* ln1_b  = (const float*)d_in[2];
    const float* qkv_w  = (const float*)d_in[3];
    const float* qkv_b  = (const float*)d_in[4];
    const float* proj_w = (const float*)d_in[5];
    const float* proj_b = (const float*)d_in[6];
    const float* ln2_g  = (const float*)d_in[7];
    const float* ln2_b  = (const float*)d_in[8];
    const float* fc1_w  = (const float*)d_in[9];
    const float* fc1_b  = (const float*)d_in[10];
    const float* fc2_w  = (const float*)d_in[11];
    const float* fc2_b  = (const float*)d_in[12];
    float* out = (float*)d_out;

    const size_t MB = 1024 * 1024;
    char* ws = (char*)d_ws;
    unsigned short* h1    = (unsigned short*)(ws + 0);          //  8 MB: LN1(x) bf16
    unsigned short* WtQKV = (unsigned short*)(ws + 8  * MB);    //  6 MB
    unsigned short* Qb    = (unsigned short*)(ws + 14 * MB);    //  8 MB
    unsigned short* Kb    = (unsigned short*)(ws + 22 * MB);    //  8 MB
    unsigned short* Vtb   = (unsigned short*)(ws + 30 * MB);    //  8 MB
    unsigned short* Yb    = (unsigned short*)(ws + 38 * MB);    //  8 MB
    unsigned short* WtP   = (unsigned short*)(ws + 46 * MB);    //  2 MB
    float*          x1    = (float*)        (ws + 48 * MB);     // 16 MB
    unsigned short* h2    = (unsigned short*)(ws + 64 * MB);    //  8 MB
    unsigned short* Wt1   = (unsigned short*)(ws + 72 * MB);    //  8 MB
    unsigned short* gact  = (unsigned short*)(ws + 80 * MB);    // 32 MB
    unsigned short* Wt2   = (unsigned short*)(ws + 112 * MB);   //  8 MB

    // weight transpose-converts
    cvtT_kernel<<<(CDIM * 3 * CDIM + 255) / 256, 256, 0, stream>>>(qkv_w,  WtQKV, CDIM, 3 * CDIM);
    cvtT_kernel<<<(CDIM * CDIM + 255) / 256,     256, 0, stream>>>(proj_w, WtP,   CDIM, CDIM);
    cvtT_kernel<<<(CDIM * MLPD + 255) / 256,     256, 0, stream>>>(fc1_w,  Wt1,   CDIM, MLPD);
    cvtT_kernel<<<(MLPD * CDIM + 255) / 256,     256, 0, stream>>>(fc2_w,  Wt2,   MLPD, CDIM);

    // LN1
    ln_kernel<<<MROWS, 256, 0, stream>>>(x, ln1_g, ln1_b, h1);

    // QKV GEMM with Q/K/V^T scatter epilogue
    gemm_bf16_kernel<0><<<dim3((3 * CDIM) / 128, MROWS / 128), 256, 0, stream>>>(
        h1, WtQKV, qkv_b, nullptr, nullptr, nullptr, Qb, Kb, Vtb, MROWS, 3 * CDIM, CDIM);

    // attention
    attn_kernel<<<BATCH * HEADS * (SEQ / 128), 256, 0, stream>>>(Qb, Kb, Vtb, Yb);

    // proj + residual(x) -> x1 (f32)
    gemm_bf16_kernel<1><<<dim3(CDIM / 128, MROWS / 128), 256, 0, stream>>>(
        Yb, WtP, proj_b, x, x1, nullptr, nullptr, nullptr, nullptr, MROWS, CDIM, CDIM);

    // LN2
    ln_kernel<<<MROWS, 256, 0, stream>>>(x1, ln2_g, ln2_b, h2);

    // FC1 + GELU -> bf16
    gemm_bf16_kernel<2><<<dim3(MLPD / 128, MROWS / 128), 256, 0, stream>>>(
        h2, Wt1, fc1_b, nullptr, nullptr, gact, nullptr, nullptr, nullptr, MROWS, MLPD, CDIM);

    // FC2 + residual(x1) -> out (f32)
    gemm_bf16_kernel<1><<<dim3(CDIM / 128, MROWS / 128), 256, 0, stream>>>(
        gact, Wt2, fc2_b, x1, out, nullptr, nullptr, nullptr, nullptr, MROWS, CDIM, MLPD);
}
